// SableNetwork_71262097375686
// MI455X (gfx1250) — matrix operand order repaired and checked by reading, not compiled, not run on hardware
//
#include <hip/hip_runtime.h>

#define B_ 16
#define S_ 512
#define E_ 512
#define H_ 8
#define L_ 3
#define D_ 64
#define OBS_ 128
#define ACT_ 16
#define RE_ (B_ * S_)
#define INV_SQRT_D_ 0.125f

typedef __attribute__((ext_vector_type(16))) __bf16 v16bf;
typedef __attribute__((ext_vector_type(8)))  float  v8f;

union Frag { v16bf v; uint4 u[2]; };

__device__ __forceinline__ v8f zero8() {
  v8f z = {0.f, 0.f, 0.f, 0.f, 0.f, 0.f, 0.f, 0.f};
  return z;
}

__device__ __forceinline__ v8f wmma_bf16f32(v16bf a, v16bf b, v8f c) {
  // D = A(16x32 bf16) x B(32x16 bf16) + C(16x16 f32)
  return __builtin_amdgcn_wmma_f32_16x16x32_bf16(false, a, false, b, (short)0, c, false, false);
}

__device__ __forceinline__ __bf16 f2bf(float f) {
  unsigned u = __builtin_bit_cast(unsigned, f);
  unsigned r = (u + 0x7fffu + ((u >> 16) & 1u)) >> 16;
  unsigned short s = (unsigned short)r;
  return __builtin_bit_cast(__bf16, s);
}

__device__ __forceinline__ float geluf(float x) {
  return 0.5f * x * (1.0f + tanhf(0.7978845608028654f * (x + 0.044715f * x * x * x)));
}

// CDNA5 async global->LDS copy (ASYNCcnt path, no VGPR bounce).
// lds is the 32-bit LDS byte offset (low 32 bits of the flat shared address).
__device__ __forceinline__ void async_b128(const void* g, unsigned lds) {
  asm volatile("global_load_async_to_lds_b128 %0, %1, off" :: "v"(lds), "v"(g) : "memory");
}
__device__ __forceinline__ void wait_async0() {
  asm volatile("s_wait_asynccnt 0x0" ::: "memory");
}

// ---------------------------------------------------------------------------
// small elementwise / conversion kernels
// ---------------------------------------------------------------------------
__global__ void k_ts(const int* __restrict__ sc, float* __restrict__ ts, int n) {
  int i = blockIdx.x * 256 + threadIdx.x;
  if (i < n) ts[i] = (float)sc[i];
}

__global__ void k_cvt(const float* __restrict__ src, __bf16* __restrict__ dst, int n) {
  int i = blockIdx.x * 256 + threadIdx.x;
  if (i < n) dst[i] = f2bf(src[i]);
}

// pad columns with zeros: src (rows, cols) f32 -> dst (rows, pitch) bf16
__global__ void k_cvt_pad(const float* __restrict__ src, __bf16* __restrict__ dst,
                          int rows, int cols, int pitch) {
  int i = blockIdx.x * 256 + threadIdx.x;
  if (i >= rows * pitch) return;
  int r = i / pitch, c = i % pitch;
  dst[i] = (c < cols) ? f2bf(src[(size_t)r * cols + c]) : f2bf(0.0f);
}

// src (K,N) f32 row-major -> dst (N,K) bf16 row-major; write-coalesced
__global__ void k_transpose_cvt(const float* __restrict__ src, __bf16* __restrict__ dst,
                                int K, int N) {
  int i = blockIdx.x * 256 + threadIdx.x;
  if (i >= K * N) return;
  int n = i / K, k = i % K;
  dst[i] = f2bf(src[(size_t)k * N + n]);
}

// src (K,N) f32 -> dst (N,Kpad) bf16 zero-padded in K
__global__ void k_transpose_cvt_pad(const float* __restrict__ src, __bf16* __restrict__ dst,
                                    int K, int N, int Kpad) {
  int i = blockIdx.x * 256 + threadIdx.x;
  if (i >= N * Kpad) return;
  int n = i / Kpad, k = i % Kpad;
  dst[i] = (k < K) ? f2bf(src[(size_t)k * N + n]) : f2bf(0.0f);
}

__global__ void k_silu_mul(const float* __restrict__ g, const float* __restrict__ p,
                           float* __restrict__ out, int n) {
  int i = blockIdx.x * 256 + threadIdx.x;
  if (i >= n) return;
  float x = g[i];
  out[i] = (x / (1.0f + __expf(-x))) * p[i];
}

// out = rmsnorm(a (+ res), w); one block per row
__global__ void k_rmsnorm(const float* __restrict__ a, const float* __restrict__ res,
                          const float* __restrict__ w, float* __restrict__ out, int N) {
  __shared__ float red[256];
  const int row = blockIdx.x;
  const float* pa = a + (size_t)row * N;
  const float* pr = res ? res + (size_t)row * N : nullptr;
  float s = 0.f;
  for (int c = threadIdx.x; c < N; c += 256) {
    float v = pa[c] + (pr ? pr[c] : 0.f);
    s += v * v;
  }
  red[threadIdx.x] = s;
  __syncthreads();
  for (int o = 128; o > 0; o >>= 1) {
    if (threadIdx.x < o) red[threadIdx.x] += red[threadIdx.x + o];
    __syncthreads();
  }
  float scale = rsqrtf(red[0] / (float)N + 1e-6f);
  for (int c = threadIdx.x; c < N; c += 256) {
    float v = pa[c] + (pr ? pr[c] : 0.f);
    out[(size_t)row * N + c] = v * scale * w[c];
  }
}

// out_bf16 = x + sinusoidal_pe(ts)   (B*S, E)
__global__ void k_add_pe(const float* __restrict__ x, const float* __restrict__ tsf,
                         __bf16* __restrict__ out, int total) {
  int i = blockIdx.x * 256 + threadIdx.x;
  if (i >= total) return;
  int row = i / E_, e = i % E_;
  float t = tsf[row];
  const int half = E_ / 2;
  int j = (e < half) ? e : e - half;
  float f = __expf(-9.210340371976184f * (float)j / (float)half);
  float ang = t * f;
  float pe = (e < half) ? __sinf(ang) : __cosf(ang);
  out[i] = f2bf(x[i] + pe);
}

// split (B,S,E) f32 q/k/v into head-major bf16: qb,kb (B,H,S,D), vt (B,H,D,S); k scaled
__global__ void k_prep_qkv(const float* __restrict__ qf, const float* __restrict__ kf,
                           const float* __restrict__ vf, __bf16* __restrict__ qb,
                           __bf16* __restrict__ kb, __bf16* __restrict__ vt) {
  int i = blockIdx.x * 256 + threadIdx.x;
  if (i >= RE_ * E_) return;
  int row = i / E_, e = i % E_;
  int b = row / S_, s = row % S_;
  int h = e / D_, d = e % D_;
  size_t bh = (size_t)b * H_ + h;
  qb[(bh * S_ + s) * D_ + d] = f2bf(qf[i]);
  kb[(bh * S_ + s) * D_ + d] = f2bf(kf[i] * INV_SQRT_D_);
  vt[(bh * D_ + d) * S_ + s] = f2bf(vf[i]);
}

// transpose-convert h0 (B,L,H,D,D) layer slice to bf16 h0t (B,H,D,D) with h0t[e][d]=h0[d][e]
__global__ void k_prep_h0(const float* __restrict__ h0, __bf16* __restrict__ h0t, int layer) {
  int i = blockIdx.x * 256 + threadIdx.x;
  if (i >= B_ * H_ * D_ * D_) return;
  int e = i % D_;
  int d = (i / D_) % D_;
  int h = (i / (D_ * D_)) % H_;
  int b = i / (D_ * D_ * H_);
  size_t src = ((((size_t)b * L_ + layer) * H_ + h) * D_ + d) * D_ + e;
  h0t[(((size_t)b * H_ + h) * D_ + e) * D_ + d] = f2bf(h0[src]);
}

// ---------------------------------------------------------------------------
// WMMA GEMM: C(M,N) f32 = A(M,K) bf16 x Bt(N,K) bf16 (B pre-transposed)
// block 256 threads (8 waves, 4x2), tile 128(M) x 128(N) x 32(K)
// double-buffered LDS fed by global_load_async_to_lds_b128 (ASYNCcnt)
// ---------------------------------------------------------------------------
__global__ void k_gemm(const __bf16* __restrict__ A, const __bf16* __restrict__ Bt,
                       float* __restrict__ C, int M, int N, int K,
                       const float* __restrict__ bias, int act) {
  __shared__ __align__(16) __bf16 As[2 * 128 * 32];
  __shared__ __align__(16) __bf16 Bs[2 * 128 * 32];
  const int tid = threadIdx.x;
  const int lane = tid & 31, wave = tid >> 5;
  const int l15 = lane & 15, hi = lane >> 4;      // lane half
  const int wm = wave & 3, wn = wave >> 2;        // 4x2 wave grid: 32M x 64N per wave
  const size_t m0 = (size_t)blockIdx.x * 128;
  const size_t n0 = (size_t)blockIdx.y * 128;
  const int lr = tid >> 1;                        // tile row 0..127 (A and B)
  const int lc = (tid & 1) * 16;                  // 0 / 16

  v8f acc[2][4];
#pragma unroll
  for (int mi = 0; mi < 2; ++mi)
#pragma unroll
    for (int ni = 0; ni < 4; ++ni) acc[mi][ni] = zero8();

  auto stage = [&](int k0, int bufi) {
    const __bf16* ag = A  + (m0 + lr) * K + k0 + lc;
    const __bf16* bg = Bt + (n0 + lr) * K + k0 + lc;
    unsigned la = (unsigned)(size_t)(As + bufi * 4096 + lr * 32 + lc);
    unsigned lb = (unsigned)(size_t)(Bs + bufi * 4096 + lr * 32 + lc);
    async_b128(ag,     la);
    async_b128(ag + 8, la + 16);
    async_b128(bg,     lb);
    async_b128(bg + 8, lb + 16);
  };

  stage(0, 0);
  wait_async0();
  __syncthreads();

  int buf = 0;
  for (int k0 = 0; k0 < K; k0 += 32) {
    if (k0 + 32 < K) {
      stage(k0 + 32, buf ^ 1);
      if (k0 + 64 < K) __builtin_prefetch(A + (m0 + lr) * K + k0 + 64, 0, 1);
    }
    const __bf16* Ab = As + buf * 4096;
    const __bf16* Bb = Bs + buf * 4096;
    Frag fa[2], fb[4];
#pragma unroll
    for (int mi = 0; mi < 2; ++mi) {
      const __bf16* arow = Ab + (wm * 32 + mi * 16 + l15) * 32;
      // 16-bit A-frag: lo lanes K{0..7,16..23}, hi lanes K{8..15,24..31}
      fa[mi].u[0] = *(const uint4*)(arow + hi * 8);
      fa[mi].u[1] = *(const uint4*)(arow + 16 + hi * 8);
    }
#pragma unroll
    for (int ni = 0; ni < 4; ++ni) {
      const __bf16* brow = Bb + (wn * 64 + ni * 16 + l15) * 32;
      // 16-bit B-frag: lo lanes K{0..7,16..23}, hi lanes K{8..15,24..31}
      fb[ni].u[0] = *(const uint4*)(brow + hi * 8);
      fb[ni].u[1] = *(const uint4*)(brow + 16 + hi * 8);
    }
#pragma unroll
    for (int mi = 0; mi < 2; ++mi)
#pragma unroll
      for (int ni = 0; ni < 4; ++ni)
        acc[mi][ni] = wmma_bf16f32(fa[mi].v, fb[ni].v, acc[mi][ni]);
    wait_async0();     // next stage's async LDS writes have landed
    __syncthreads();
    buf ^= 1;
  }

#pragma unroll
  for (int mi = 0; mi < 2; ++mi)
#pragma unroll
    for (int ni = 0; ni < 4; ++ni)
#pragma unroll
      for (int q = 0; q < 8; ++q) {
        size_t row = m0 + wm * 32 + mi * 16 + q + hi * 8;  // C layout: M = vgpr + 8*laneHi
        size_t col = n0 + wn * 64 + ni * 16 + l15;         // N = lane&15
        float v = acc[mi][ni][q];
        if (bias) v += bias[col];
        if (act == 1) v = geluf(v);
        C[row * N + col] = v;
      }
}

// ---------------------------------------------------------------------------
// Retention core: one block = (b, h, 64 query rows), 4 waves x 16-row tiles.
// scores = (q kT) .* decay  -> bf16 via LDS transpose -> scores @ v  (+ q@h0 * xi)
// ---------------------------------------------------------------------------
__global__ void k_retention(const __bf16* __restrict__ qb, const __bf16* __restrict__ kb,
                            const __bf16* __restrict__ vt, const __bf16* __restrict__ h0t,
                            const float* __restrict__ tsf, float* __restrict__ out,
                            int masked) {
  __shared__ __align__(16) __bf16 sc[4 * 16 * 40];   // per-wave 16x32 score tile (padded)
  const int b = blockIdx.z, h = blockIdx.y;
  const int wave = threadIdx.x >> 5, lane = threadIdx.x & 31;
  const int l15 = lane & 15, hi = lane >> 4;
  const int n0 = blockIdx.x * 64 + wave * 16;
  const float kappa = 1.0f - exp2f(-5.0f - (float)h);
  const float lk = logf(kappa);
  const size_t bh = (size_t)b * H_ + h;

  // q A-fragments for this 16-row tile (K = D = 64 -> two 16x32 frags)
  const __bf16* qrow = qb + (bh * S_ + n0 + l15) * D_;
  Frag qa1, qa2;
  qa1.u[0] = *(const uint4*)(qrow + hi * 8);
  qa1.u[1] = *(const uint4*)(qrow + 16 + hi * 8);
  qa2.u[0] = *(const uint4*)(qrow + 32 + hi * 8);
  qa2.u[1] = *(const uint4*)(qrow + 48 + hi * 8);

  float tsn[8];
#pragma unroll
  for (int r = 0; r < 8; ++r) tsn[r] = tsf[(size_t)b * S_ + n0 + r + hi * 8];

  // cross-chunk term: (q @ h0) * xi  seeds the output accumulators
  v8f acc[4];
#pragma unroll
  for (int et = 0; et < 4; ++et) {
    const __bf16* hrow = h0t + (bh * D_ + et * 16 + l15) * D_;
    Frag hb1, hb2;
    hb1.u[0] = *(const uint4*)(hrow + hi * 8);
    hb1.u[1] = *(const uint4*)(hrow + 16 + hi * 8);
    hb2.u[0] = *(const uint4*)(hrow + 32 + hi * 8);
    hb2.u[1] = *(const uint4*)(hrow + 48 + hi * 8);
    v8f z = zero8();
    z = wmma_bf16f32(qa1.v, hb1.v, z);
    z = wmma_bf16f32(qa2.v, hb2.v, z);
#pragma unroll
    for (int r = 0; r < 8; ++r) z[r] *= __expf(lk * (tsn[r] + 1.0f));
    acc[et] = z;
  }

  __bf16* scw = sc + wave * 16 * 40;

  for (int mp = 0; mp < S_ / 32; ++mp) {
#pragma unroll
    for (int half = 0; half < 2; ++half) {
      const int mt0 = mp * 32 + half * 16;
      const __bf16* krow = kb + (bh * S_ + mt0 + l15) * D_;
      Frag kf1, kf2;
      kf1.u[0] = *(const uint4*)(krow + hi * 8);
      kf1.u[1] = *(const uint4*)(krow + 16 + hi * 8);
      kf2.u[0] = *(const uint4*)(krow + 32 + hi * 8);
      kf2.u[1] = *(const uint4*)(krow + 48 + hi * 8);
      v8f s = zero8();
      s = wmma_bf16f32(qa1.v, kf1.v, s);
      s = wmma_bf16f32(qa2.v, kf2.v, s);
      const int m = mt0 + l15;                       // lane owns one key column
      const float tsm = tsf[(size_t)b * S_ + m];
#pragma unroll
      for (int r = 0; r < 8; ++r) {
        const int n = n0 + r + hi * 8;
        const float diff = tsn[r] - tsm;
        float dm;
        if (masked) {
          bool allow = (diff > 0.f) || ((diff == 0.f) && (n >= m));
          dm = allow ? __expf(lk * diff) : 0.f;
        } else {
          dm = __expf(lk * fabsf(diff));
        }
        scw[(r + hi * 8) * 40 + half * 16 + l15] = f2bf(s[r] * dm);
      }
    }
    __syncthreads();
    // A-frag of the transposed 16x32 score tile
    Frag sa;
    sa.u[0] = *(const uint4*)(scw + l15 * 40 + hi * 8);
    sa.u[1] = *(const uint4*)(scw + l15 * 40 + 16 + hi * 8);
#pragma unroll
    for (int et = 0; et < 4; ++et) {
      const __bf16* vrow = vt + (bh * D_ + et * 16 + l15) * S_ + mp * 32;
      Frag vb;
      vb.u[0] = *(const uint4*)(vrow + hi * 8);
      vb.u[1] = *(const uint4*)(vrow + 16 + hi * 8);
      acc[et] = wmma_bf16f32(sa.v, vb.v, acc[et]);
    }
    __syncthreads();
  }

#pragma unroll
  for (int et = 0; et < 4; ++et)
#pragma unroll
    for (int r = 0; r < 8; ++r) {
      const int n = n0 + r + hi * 8;
      out[((size_t)b * S_ + n) * E_ + h * D_ + et * 16 + l15] = acc[et][r];
    }
}

// ---------------------------------------------------------------------------
// state update: h_new = exp(lk*(tmax+1))*h0 + sum_m wst[m] * k[m,:]^T v[m,:]
// one block per (h,b); kf unscaled f32 (scale folded in at the end)
// ---------------------------------------------------------------------------
__global__ void k_hnew(const float* __restrict__ kf, const float* __restrict__ vf,
                       const float* __restrict__ h0, const float* __restrict__ tsf,
                       float* __restrict__ hout, int layer) {
  const int h = blockIdx.x, b = blockIdx.y;
  const int e = threadIdx.x & 63, dg = threadIdx.x >> 6;
  const float kappa = 1.0f - exp2f(-5.0f - (float)h);
  const float lk = logf(kappa);
  float tmax = -1e30f;
  for (int m = 0; m < S_; ++m) tmax = fmaxf(tmax, tsf[(size_t)b * S_ + m]);
  float accv[16];
#pragma unroll
  for (int i = 0; i < 16; ++i) accv[i] = 0.f;
  for (int m = 0; m < S_; ++m) {
    const float wst = __expf(lk * (tmax - tsf[(size_t)b * S_ + m]));
    const float* rowp = kf + ((size_t)b * S_ + m) * E_ + h * D_ + dg * 16;
    const float vv = wst * vf[((size_t)b * S_ + m) * E_ + h * D_ + e];
#pragma unroll
    for (int i = 0; i < 16; ++i) accv[i] += rowp[i] * vv;
  }
  const float dec = __expf(lk * (tmax + 1.0f));
  const size_t base = (((size_t)b * L_ + layer) * H_ + h) * D_ * D_;
#pragma unroll
  for (int i = 0; i < 16; ++i) {
    const int d = dg * 16 + i;
    hout[base + (size_t)d * D_ + e] = dec * h0[base + (size_t)d * D_ + e] + INV_SQRT_D_ * accv[i];
  }
}

// small output heads: out(R,Nout) = x(R,K) @ w(K,Nout) + b
__global__ void k_head(const float* __restrict__ x, const float* __restrict__ w,
                       const float* __restrict__ bias, float* __restrict__ out,
                       int R, int K, int Nout) {
  int t = blockIdx.x * 256 + threadIdx.x;
  if (t >= R * Nout) return;
  int row = t / Nout, n = t % Nout;
  float s = bias[n];
  const float* xr = x + (size_t)row * K;
  for (int e = 0; e < K; ++e) s += xr[e] * w[(size_t)e * Nout + n];
  out[t] = s;
}

// ---------------------------------------------------------------------------
// host orchestration
// ---------------------------------------------------------------------------
extern "C" void kernel_launch(void* const* d_in, const int* in_sizes, int n_in,
                              void* d_out, int out_size, void* d_ws, size_t ws_size,
                              hipStream_t stream) {
  (void)in_sizes; (void)n_in; (void)out_size; (void)ws_size;

  const float* obs     = (const float*)d_in[0];
  const float* action  = (const float*)d_in[1];
  const float* enc_hs  = (const float*)d_in[2];
  const float* dec_hs1 = (const float*)d_in[3];
  const float* dec_hs2 = (const float*)d_in[4];
  const int*   stepc   = (const int*)d_in[5];
  const float* eoln = (const float*)d_in[6];
  const float* eow  = (const float*)d_in[7];
  const float* eln  = (const float*)d_in[8];
  const float* ewq  = (const float*)d_in[9];
  const float* ewk  = (const float*)d_in[10];
  const float* ewv  = (const float*)d_in[11];
  const float* ewo  = (const float*)d_in[12];
  const float* eln1 = (const float*)d_in[13];
  const float* eln2 = (const float*)d_in[14];
  const float* efg  = (const float*)d_in[15];
  const float* efp  = (const float*)d_in[16];
  const float* ehw1 = (const float*)d_in[17];
  const float* ehb1 = (const float*)d_in[18];
  const float* ehln = (const float*)d_in[19];
  const float* ehw2 = (const float*)d_in[20];
  const float* ehb2 = (const float*)d_in[21];
  const float* daw  = (const float*)d_in[22];
  const float* dln  = (const float*)d_in[23];
  const float* dwq1 = (const float*)d_in[24];
  const float* dwk1 = (const float*)d_in[25];
  const float* dwv1 = (const float*)d_in[26];
  const float* dwo1 = (const float*)d_in[27];
  const float* dwq2 = (const float*)d_in[28];
  const float* dwk2 = (const float*)d_in[29];
  const float* dwv2 = (const float*)d_in[30];
  const float* dwo2 = (const float*)d_in[31];
  const float* dln1 = (const float*)d_in[32];
  const float* dln2 = (const float*)d_in[33];
  const float* dln3 = (const float*)d_in[34];
  const float* dfg  = (const float*)d_in[35];
  const float* dfp  = (const float*)d_in[36];
  const float* dhw1 = (const float*)d_in[37];
  const float* dhb1 = (const float*)d_in[38];
  const float* dhln = (const float*)d_in[39];
  const float* dhw2 = (const float*)d_in[40];
  const float* dhb2 = (const float*)d_in[41];

  const size_t HSN = (size_t)B_ * L_ * H_ * D_ * D_;
  float* out_value  = (float*)d_out;
  float* out_logits = out_value + RE_;
  float* out_ench   = out_logits + (size_t)RE_ * ACT_;
  float* out_d1     = out_ench + HSN;
  float* out_d2     = out_d1 + HSN;

  // --- workspace bump allocator ---
  char* ws = (char*)d_ws;
  size_t cur = 0;
  auto alloc = [&](size_t bytes) -> void* {
    void* p = ws + cur;
    cur = (cur + bytes + 255) & ~(size_t)255;
    return p;
  };
  const size_t FE = (size_t)RE_ * E_;            // elems in a (B,S,E) tensor
  auto fbuf = [&]() { return (float*)alloc(FE * 4); };
  auto bbuf = [&]() { return (__bf16*)alloc(FE * 2); };

  float* tsf     = (float*)alloc((size_t)RE_ * 4);
  float* obs_rep = fbuf();
  float* xstate  = fbuf();
  float* xn      = fbuf();
  float* xr      = fbuf();
  float* qf      = fbuf();
  float* kf      = fbuf();
  float* vf      = fbuf();
  float* retf    = fbuf();
  float* gf      = fbuf();
  float* pf      = fbuf();
  float* obuf    = fbuf();
  __bf16* peb  = bbuf();
  __bf16* opeb = bbuf();
  __bf16* xb   = bbuf();
  __bf16* retb = bbuf();
  __bf16* qb   = bbuf();
  __bf16* kb   = bbuf();
  __bf16* vtb  = bbuf();
  __bf16* h0t  = (__bf16*)alloc((size_t)B_ * H_ * D_ * D_ * 2);
  __bf16* obsb = (__bf16*)alloc((size_t)RE_ * OBS_ * 2);
  __bf16* actb = (__bf16*)alloc((size_t)RE_ * 32 * 2);
  __bf16* wTq  = (__bf16*)alloc((size_t)E_ * E_ * 2);
  __bf16* wTk  = (__bf16*)alloc((size_t)E_ * E_ * 2);
  __bf16* wTv  = (__bf16*)alloc((size_t)E_ * E_ * 2);
  __bf16* wTo  = (__bf16*)alloc((size_t)E_ * E_ * 2);
  __bf16* wT0  = (__bf16*)alloc((size_t)E_ * E_ * 2);
  __bf16* wT1  = (__bf16*)alloc((size_t)E_ * E_ * 2);
  __bf16* dawT = (__bf16*)alloc((size_t)E_ * 32 * 2);

  const int EE = E_ * E_;
  auto nblk = [](size_t n) { return (unsigned)((n + 255) / 256); };
  auto tcvt = [&](const float* src, __bf16* dst, int K, int N) {
    k_transpose_cvt<<<nblk((size_t)K * N), 256, 0, stream>>>(src, dst, K, N);
  };
  auto cvt = [&](const float* src, __bf16* dst, size_t n) {
    k_cvt<<<nblk(n), 256, 0, stream>>>(src, dst, (int)n);
  };
  auto gemm = [&](const __bf16* A, const __bf16* Bt, float* C, int M, int N, int K,
                  const float* bias, int act) {
    dim3 g(M / 128, N / 128);
    k_gemm<<<g, 256, 0, stream>>>(A, Bt, C, M, N, K, bias, act);
  };
  auto rms = [&](const float* a, const float* res, const float* w, float* o, int N) {
    k_rmsnorm<<<RE_, 256, 0, stream>>>(a, res, w, o, N);
  };
  auto pe = [&](const float* x, __bf16* o) {
    k_add_pe<<<nblk(FE), 256, 0, stream>>>(x, tsf, o, (int)FE);
  };
  // full retention: projections -> core -> state update -> output projection
  auto do_ret = [&](const __bf16* xqb, const __bf16* xkb, const __bf16* xvb,
                    const float* wq, const float* wk, const float* wv, const float* wo,
                    const float* hsrc, float* hdst, int layer, int masked, float* oout) {
    tcvt(wq, wTq, E_, E_);
    tcvt(wk, wTk, E_, E_);
    tcvt(wv, wTv, E_, E_);
    tcvt(wo, wTo, E_, E_);
    gemm(xqb, wTq, qf, RE_, E_, E_, nullptr, 0);
    gemm(xkb, wTk, kf, RE_, E_, E_, nullptr, 0);
    gemm(xvb, wTv, vf, RE_, E_, E_, nullptr, 0);
    k_prep_qkv<<<nblk(FE), 256, 0, stream>>>(qf, kf, vf, qb, kb, vtb);
    k_prep_h0<<<nblk((size_t)B_ * H_ * D_ * D_), 256, 0, stream>>>(hsrc, h0t, layer);
    k_retention<<<dim3(S_ / 64, H_, B_), 128, 0, stream>>>(qb, kb, vtb, h0t, tsf, retf, masked);
    k_hnew<<<dim3(H_, B_), 256, 0, stream>>>(kf, vf, hsrc, tsf, hdst, layer);
    cvt(retf, retb, FE);
    gemm(retb, wTo, oout, RE_, E_, E_, nullptr, 0);
  };

  // ---- timesteps ----
  k_ts<<<nblk(RE_), 256, 0, stream>>>(stepc, tsf, RE_);

  // ---- encoder input projection: obs_rep = gelu(rmsnorm(obs) @ eow) ----
  k_rmsnorm<<<RE_, 256, 0, stream>>>(obs, nullptr, eoln, xn, OBS_);
  cvt(xn, obsb, (size_t)RE_ * OBS_);
  tcvt(eow, wT0, OBS_, E_);
  gemm(obsb, wT0, obs_rep, RE_, E_, OBS_, nullptr, 1);

  // ---- encoder layers ----
  for (int i = 0; i < L_; ++i) {
    rms(obs_rep, nullptr, eln, xn, E_);
    pe(xn, peb);
    do_ret(peb, peb, peb, ewq + (size_t)i * EE, ewk + (size_t)i * EE,
           ewv + (size_t)i * EE, ewo + (size_t)i * EE, enc_hs, out_ench, i, 0, obuf);
    rms(xn, obuf, eln1 + (size_t)i * E_, xr, E_);
    cvt(xr, xb, FE);
    tcvt(efg + (size_t)i * EE, wT0, E_, E_);
    tcvt(efp + (size_t)i * EE, wT1, E_, E_);
    gemm(xb, wT0, gf, RE_, E_, E_, nullptr, 0);
    gemm(xb, wT1, pf, RE_, E_, E_, nullptr, 0);
    k_silu_mul<<<nblk(FE), 256, 0, stream>>>(gf, pf, retf, (int)FE);
    rms(xr, retf, eln2 + (size_t)i * E_, obs_rep, E_);
  }

  // ---- value head ----
  cvt(obs_rep, xb, FE);
  tcvt(ehw1, wT0, E_, E_);
  gemm(xb, wT0, gf, RE_, E_, E_, ehb1, 1);
  rms(gf, nullptr, ehln, pf, E_);
  k_head<<<nblk((size_t)RE_), 256, 0, stream>>>(pf, ehw2, ehb2, out_value, RE_, E_, 1);

  // ---- decoder input: x = rmsnorm(gelu(action @ daw)) ----
  k_cvt_pad<<<nblk((size_t)RE_ * 32), 256, 0, stream>>>(action, actb, RE_, ACT_ + 1, 32);
  k_transpose_cvt_pad<<<nblk((size_t)E_ * 32), 256, 0, stream>>>(daw, dawT, ACT_ + 1, E_, 32);
  gemm(actb, dawT, xn, RE_, E_, 32, nullptr, 1);
  rms(xn, nullptr, dln, xstate, E_);
  pe(obs_rep, opeb);  // obs_rep + pe is reused by every decoder cross-retention

  // ---- decoder layers ----
  for (int i = 0; i < L_; ++i) {
    pe(xstate, peb);
    do_ret(peb, peb, peb, dwq1 + (size_t)i * EE, dwk1 + (size_t)i * EE,
           dwv1 + (size_t)i * EE, dwo1 + (size_t)i * EE, dec_hs1, out_d1, i, 1, obuf);
    rms(xstate, obuf, dln1 + (size_t)i * E_, xr, E_);
    // cross retention: xq = obs_rep(+pe), xk = xv = r(+pe)
    pe(xr, peb);
    do_ret(opeb, peb, peb, dwq2 + (size_t)i * EE, dwk2 + (size_t)i * EE,
           dwv2 + (size_t)i * EE, dwo2 + (size_t)i * EE, dec_hs2, out_d2, i, 1, obuf);
    rms(obs_rep, obuf, dln2 + (size_t)i * E_, xn, E_);
    cvt(xn, xb, FE);
    tcvt(dfg + (size_t)i * EE, wT0, E_, E_);
    tcvt(dfp + (size_t)i * EE, wT1, E_, E_);
    gemm(xb, wT0, gf, RE_, E_, E_, nullptr, 0);
    gemm(xb, wT1, pf, RE_, E_, E_, nullptr, 0);
    k_silu_mul<<<nblk(FE), 256, 0, stream>>>(gf, pf, retf, (int)FE);
    rms(xn, retf, dln3 + (size_t)i * E_, xstate, E_);
  }

  // ---- logits head ----
  cvt(xstate, xb, FE);
  tcvt(dhw1, wT0, E_, E_);
  gemm(xb, wT0, gf, RE_, E_, E_, dhb1, 1);
  rms(gf, nullptr, dhln, pf, E_);
  k_head<<<nblk((size_t)RE_ * ACT_), 256, 0, stream>>>(pf, dhw2, dhb2, out_logits, RE_, E_, ACT_);
}
